// DNCModule_88261577933100
// MI455X (gfx1250) — matrix operand order repaired and checked by reading, not compiled, not run on hardware
//
#include <hip/hip_runtime.h>
#include <hip/hip_bf16.h>

// ---------------------------------------------------------------------------
// DNC forward for MI455X (gfx1250, wave32, WMMA).
// GEMMs use v_wmma_f32_16x16x32_bf16; the dominant gates GEMM stages its A
// panel into LDS via GLOBAL_LOAD_ASYNC_TO_LDS_B128 (ASYNCcnt) and computes
// 16x64 per wave; content-addressing sweep fuses WMMA dot products with
// per-slot norms; write/read use K=8 sparse top-k with 8-slot score fix-up.
// ---------------------------------------------------------------------------

typedef __attribute__((ext_vector_type(16))) __bf16 v16bf;
typedef __attribute__((ext_vector_type(8)))  float  v8f;
typedef int vi4 __attribute__((vector_size(16)));   // async-LDS builtin operand type

#ifndef __has_builtin
#define __has_builtin(x) 0
#endif
#if __has_builtin(__builtin_amdgcn_global_load_async_to_lds_b128)
#define HAVE_ASYNC_LDS 1
#else
#define HAVE_ASYNC_LDS 0
#endif
#if __has_builtin(__builtin_amdgcn_s_wait_asynccnt)
#define HAVE_WAIT_ASYNC 1
#else
#define HAVE_WAIT_ASYNC 0
#endif

#define B_    128
#define T_    32
#define D_    256
#define H_    512
#define S_    2048
#define M_    64
#define R_    4
#define OUT_  256
#define KIN_  768      // D_ + H_
#define G4H_  2048     // 4*H_
#define CTRLN_ 709     // OUT + R*M + R + 3*W*M + W
#define N2_   720      // padded vt|ctrl width
#define KTOP_ 8
#define EPS_  1e-8f
#define SA_LD (KIN_ + 8)   // LDS row pitch in bf16: 1552B -> bank step 4, no conflicts

__device__ __forceinline__ float sigf(float x)      { return 1.f / (1.f + __expf(-x)); }
__device__ __forceinline__ float softplusf(float x) { return x > 20.f ? x : log1pf(__expf(x)); }

// ---------------------------------------------------------------------------
// Generic WMMA GEMM:  C[M,N] = A[M,K](bf16,row-major) * Bw[N,K]^T (bf16)
//                     (+ bias[N]) (+ addend[M,ldadd]);  16x16 tile per wave.
// ---------------------------------------------------------------------------
__global__ void wmma_gemm_nt(const __bf16* __restrict__ A, int lda,
                             const __bf16* __restrict__ Bw, int ldb,
                             const float* __restrict__ bias,
                             const float* __restrict__ addend, int ldadd,
                             float* __restrict__ C, int ldc,
                             int Mdim, int Ndim, int Kdim)
{
    const int lane = threadIdx.x & 31;
    const int wave = threadIdx.x >> 5;
    const int mtiles = (Mdim + 15) >> 4;
    const int ntiles = (Ndim + 15) >> 4;
    const int tile = blockIdx.x * (blockDim.x >> 5) + wave;
    if (tile >= mtiles * ntiles) return;
    const int mT = (tile / ntiles) * 16;
    const int nT = (tile % ntiles) * 16;
    const int row = lane & 15;
    const int hi  = lane >> 4;

    v8f acc = {};
    for (int kb = 0; kb < Kdim; kb += 32) {
        v16bf a, b;
        const __bf16* ap = A + (size_t)(mT + row) * lda + kb + hi * 16;
#pragma unroll
        for (int i = 0; i < 16; ++i) a[i] = ap[i];
        const __bf16* bp = Bw + (size_t)(nT + row) * ldb + kb + hi * 16;
#pragma unroll
        for (int i = 0; i < 16; ++i) b[i] = bp[i];
        acc = __builtin_amdgcn_wmma_f32_16x16x32_bf16(false, a, false, b,
                                                      (short)0, acc, false, false);
    }
    const int n = nT + row;
    if (n >= Ndim) return;
#pragma unroll
    for (int v = 0; v < 8; ++v) {
        int m = mT + v + hi * 8;
        if (m < Mdim) {
            float r = acc[v];
            if (bias)   r += bias[n];
            if (addend) r += addend[(size_t)m * ldadd + n];
            C[(size_t)m * ldc + n] = r;
        }
    }
}

// ---------------------------------------------------------------------------
// LSTM-gates GEMM: C[128,2048] = xh[128,768] @ Wcat[2048,768]^T + bsum.
// Block = 256 thr (8 waves).  Block stages one 16x768 A panel into LDS with
// async global->LDS b128 copies (ASYNCcnt), then each wave computes a 16x64
// tile (4 WMMA accumulators), reusing the LDS A fragment 4x per K step.
// Grid: 8 m-tiles x 4 n-groups = 32 blocks.
// ---------------------------------------------------------------------------
__global__ void k_gates_gemm(const __bf16* __restrict__ A,
                             const __bf16* __restrict__ Bw,
                             const float* __restrict__ bias,
                             float* __restrict__ C)
{
    __shared__ __align__(16) __bf16 sA[16 * SA_LD];
    const int tid  = threadIdx.x;
    const int lane = tid & 31;
    const int wave = tid >> 5;
    const int mT    = (blockIdx.x >> 2) * 16;
    const int nBase = (blockIdx.x & 3) * 512 + wave * 64;

    // ---- stage A panel: 16 rows x 768 bf16 = 1536 x 16B chunks (96/row) ----
    {
#if HAVE_ASYNC_LDS
        for (int i = tid; i < 1536; i += 256) {
            int r = i / 96, cc = i % 96;
            __builtin_amdgcn_global_load_async_to_lds_b128(
                (vi4*)(A + (size_t)(mT + r) * KIN_ + cc * 8),
                (vi4*)(sA + (size_t)r * SA_LD + cc * 8), 0, 0);
        }
#if HAVE_WAIT_ASYNC
        __builtin_amdgcn_s_wait_asynccnt(0);
#else
        asm volatile("s_wait_asynccnt 0" ::: "memory");
#endif
#else
        for (int i = tid; i < 1536; i += 256) {
            int r = i / 96, cc = i % 96;
            const int4 v = *(const int4*)(A + (size_t)(mT + r) * KIN_ + cc * 8);
            *(int4*)(sA + (size_t)r * SA_LD + cc * 8) = v;
        }
#endif
    }
    __syncthreads();

    const int row = lane & 15;
    const int hi  = lane >> 4;
    v8f acc[4] = {};
    for (int kb = 0; kb < KIN_; kb += 32) {
        v16bf a;
        const __bf16* ap = sA + (size_t)row * SA_LD + kb + hi * 16;   // ds_load
#pragma unroll
        for (int i = 0; i < 16; ++i) a[i] = ap[i];
#pragma unroll
        for (int s = 0; s < 4; ++s) {
            v16bf b;
            const __bf16* bp = Bw + (size_t)(nBase + s * 16 + row) * KIN_ + kb + hi * 16;
#pragma unroll
            for (int i = 0; i < 16; ++i) b[i] = bp[i];
            acc[s] = __builtin_amdgcn_wmma_f32_16x16x32_bf16(false, a, false, b,
                                                             (short)0, acc[s], false, false);
        }
    }
#pragma unroll
    for (int s = 0; s < 4; ++s) {
        int n = nBase + s * 16 + row;
#pragma unroll
        for (int v = 0; v < 8; ++v) {
            int m = mT + v + hi * 8;
            C[(size_t)m * G4H_ + n] = acc[s][v] + bias[n];
        }
    }
}

// ---------------------------------------------------------------------------
// Content addressing sweep (pre-write memory): one wave per (batch, 16-slot
// tile).  A = keys[b] (16x64 bf16; rows 0..4 = beta/||k||-scaled heads).
// Slot L2-norms accumulated from the same f32 loads, halves combined via
// shfl_xor(16).  scores[b,h,s] = beta_h * cos(k_h, mem_s).
// ---------------------------------------------------------------------------
__global__ void k_scores(const float* __restrict__ mem,
                         const __bf16* __restrict__ keys,
                         float* __restrict__ scores)
{
    const int lane = threadIdx.x & 31;
    const int wave = threadIdx.x >> 5;
    const int tile = blockIdx.x * (blockDim.x >> 5) + wave;
    if (tile >= B_ * (S_ / 16)) return;
    const int b  = tile / (S_ / 16);
    const int s0 = (tile % (S_ / 16)) * 16;
    const int row = lane & 15;
    const int hi  = lane >> 4;
    const float*  memb = mem  + (size_t)b * S_ * M_;
    const __bf16* kp   = keys + (size_t)b * 16 * M_;

    v8f acc = {};
    float ss = 0.f;
    for (int kk = 0; kk < M_; kk += 32) {
        v16bf a, bm;
        const __bf16* ap = kp + row * M_ + kk + hi * 16;
#pragma unroll
        for (int i = 0; i < 16; ++i) a[i] = ap[i];
        const float* mp = memb + (size_t)(s0 + row) * M_ + kk + hi * 16;
#pragma unroll
        for (int i = 0; i < 16; ++i) { float f = mp[i]; ss += f * f; bm[i] = (__bf16)f; }
        acc = __builtin_amdgcn_wmma_f32_16x16x32_bf16(false, a, false, bm,
                                                      (short)0, acc, false, false);
    }
    ss += __shfl_xor(ss, 16, 32);
    float inv = 1.f / fmaxf(__builtin_sqrtf(ss), EPS_);
    if (hi == 0) {
        float* sc = scores + (size_t)b * 5 * S_ + (s0 + row);
#pragma unroll
        for (int h = 0; h < 5; ++h) sc[(size_t)h * S_] = acc[h] * inv;
    }
}

// ---------------------------------------------------------------------------
__global__ void k_cell(const float* __restrict__ gates,
                       float* __restrict__ c, __bf16* __restrict__ xh)
{
    int idx = blockIdx.x * blockDim.x + threadIdx.x;
    if (idx >= B_ * H_) return;
    int b = idx / H_, h = idx % H_;
    const float* g = gates + (size_t)b * G4H_;
    float gi = g[h], gf = g[H_ + h], gg = g[2 * H_ + h], go = g[3 * H_ + h];
    float cn = sigf(gf) * c[idx] + sigf(gi) * tanhf(gg);
    c[idx] = cn;
    xh[(size_t)b * KIN_ + D_ + h] = (__bf16)(sigf(go) * tanhf(cn));
}

// ---------------------------------------------------------------------------
// Heads from ctrl = out2[b, 256:256+453]; keys scaled by beta/||tanh(k)||.
// ---------------------------------------------------------------------------
__global__ void k_heads(const float* __restrict__ out2,
                        __bf16* __restrict__ keys,
                        float* __restrict__ erase, float* __restrict__ addv)
{
    const int b = blockIdx.x;
    const float* ctrl = out2 + (size_t)b * N2_ + OUT_;
    __shared__ float sq[320];
    __shared__ float scale[5];
    const int tid = threadIdx.x;
    const int h = tid >> 6, d = tid & 63;

    float raw = (h == 0) ? ctrl[260 + d] : ctrl[(h - 1) * 64 + d];
    float val = tanhf(raw);
    sq[tid] = val * val;
    __syncthreads();
    if (d == 0) {
        float s = 0.f;
        for (int j = 0; j < 64; ++j) s += sq[h * 64 + j];
        float braw = (h == 0) ? ctrl[324] : ctrl[256 + (h - 1)];
        scale[h] = softplusf(braw) / fmaxf(__builtin_sqrtf(s), EPS_);
    }
    __syncthreads();
    __bf16* kb = keys + (size_t)b * 16 * M_;
    kb[h * 64 + d] = (__bf16)(val * scale[h]);
    for (int i = tid; i < 11 * 64; i += 320) kb[320 + i] = (__bf16)0.f;
    if (tid < 64) {
        erase[b * 64 + tid] = sigf(ctrl[325 + tid]);
        addv [b * 64 + tid] = tanhf(ctrl[389 + tid]);
    }
}

// ---------------------------------------------------------------------------
__device__ void softmax_top8(float* ls, float* red, int* redi,
                             float* topw, int* topi, int tid)
{
    float mx = -1e30f;
    for (int s = tid; s < S_; s += 256) mx = fmaxf(mx, ls[s]);
    red[tid] = mx; __syncthreads();
    for (int o = 128; o > 0; o >>= 1) {
        if (tid < o) red[tid] = fmaxf(red[tid], red[tid + o]);
        __syncthreads();
    }
    float m = red[0]; __syncthreads();
    float se = 0.f;
    for (int s = tid; s < S_; s += 256) se += __expf(ls[s] - m);
    red[tid] = se; __syncthreads();
    for (int o = 128; o > 0; o >>= 1) {
        if (tid < o) red[tid] += red[tid + o];
        __syncthreads();
    }
    float denom = red[0]; __syncthreads();
    for (int j = 0; j < KTOP_; ++j) {
        float bm = -1e30f; int bi = 0;
        for (int s = tid; s < S_; s += 256)
            if (ls[s] > bm) { bm = ls[s]; bi = s; }
        red[tid] = bm; redi[tid] = bi; __syncthreads();
        for (int o = 128; o > 0; o >>= 1) {
            if (tid < o && red[tid + o] > red[tid]) {
                red[tid] = red[tid + o]; redi[tid] = redi[tid + o];
            }
            __syncthreads();
        }
        if (tid == 0) {
            topi[j] = redi[0];
            topw[j] = __expf(red[0] - m) / denom;
            ls[redi[0]] = -1e30f;
        }
        __syncthreads();
    }
}

// ---------------------------------------------------------------------------
__global__ void k_write(float* __restrict__ mem, float* __restrict__ scores,
                        const __bf16* __restrict__ keys,
                        const float* __restrict__ erase,
                        const float* __restrict__ addv)
{
    const int b = blockIdx.x, tid = threadIdx.x;
    __shared__ float ls[S_];
    __shared__ float red[256];
    __shared__ int   redi[256];
    __shared__ float topw[KTOP_];
    __shared__ int   topi[KTOP_];
    __shared__ float nrmI[KTOP_];

    float* wsc = scores + (size_t)b * 5 * S_;
    for (int s = tid; s < S_; s += 256) ls[s] = wsc[s];
    __syncthreads();
    softmax_top8(ls, red, redi, topw, topi, tid);

    if (tid < KTOP_ * M_) {
        int j = tid >> 6, d = tid & 63;
        float w = topw[j];
        float* mp = mem + ((size_t)b * S_ + topi[j]) * M_ + d;
        *mp = *mp * (1.f - w * erase[b * 64 + d]) + w * addv[b * 64 + d];
    }
    __syncthreads();

    if (tid < KTOP_) {
        const float* mp = mem + ((size_t)b * S_ + topi[tid]) * M_;
        float ss = 0.f;
        for (int d = 0; d < M_; ++d) { float f = mp[d]; ss += f * f; }
        nrmI[tid] = 1.f / fmaxf(__builtin_sqrtf(ss), EPS_);
    }
    __syncthreads();

    if (tid < KTOP_ * R_) {
        int j = tid >> 2, r = tid & 3, s = topi[j];
        const float*  mp = mem  + ((size_t)b * S_ + s) * M_;
        const __bf16* kp = keys + (size_t)b * 16 * M_ + (1 + r) * M_;
        float dot = 0.f;
        for (int d = 0; d < M_; ++d) dot += (float)kp[d] * mp[d];
        scores[(size_t)b * 5 * S_ + (size_t)(1 + r) * S_ + s] = dot * nrmI[j];
    }
}

// ---------------------------------------------------------------------------
__global__ void k_read(const float* __restrict__ mem, float* __restrict__ scores,
                       __bf16* __restrict__ readsb)
{
    const int b = blockIdx.x >> 2, r = blockIdx.x & 3, tid = threadIdx.x;
    __shared__ float ls[S_];
    __shared__ float red[256];
    __shared__ int   redi[256];
    __shared__ float topw[KTOP_];
    __shared__ int   topi[KTOP_];

    const float* rsc = scores + (size_t)b * 5 * S_ + (size_t)(1 + r) * S_;
    for (int s = tid; s < S_; s += 256) ls[s] = rsc[s];
    __syncthreads();
    softmax_top8(ls, red, redi, topw, topi, tid);

    if (tid < M_) {
        float acc = 0.f;
        for (int j = 0; j < KTOP_; ++j)
            acc += topw[j] * mem[((size_t)b * S_ + topi[j]) * M_ + tid];
        readsb[(size_t)b * (R_ * M_) + r * M_ + tid] = (__bf16)acc;
    }
}

// --------------------------- prep / glue kernels ---------------------------
__global__ void k_cvt_wcat(const float* __restrict__ Wih, const float* __restrict__ Whh,
                           __bf16* __restrict__ Wcat)
{
    int idx = blockIdx.x * blockDim.x + threadIdx.x;
    if (idx >= G4H_ * KIN_) return;
    int n = idx / KIN_, k = idx % KIN_;
    float v = (k < D_) ? Wih[(size_t)n * D_ + k] : Whh[(size_t)n * H_ + (k - D_)];
    Wcat[idx] = (__bf16)v;
}

__global__ void k_cvt_wo2(const float* __restrict__ Wout, const float* __restrict__ Whead,
                          __bf16* __restrict__ Wo2)
{
    int idx = blockIdx.x * blockDim.x + threadIdx.x;
    if (idx >= N2_ * H_) return;
    int n = idx / H_, k = idx % H_;
    float v = (n < OUT_) ? Wout[(size_t)n * H_ + k]
            : (n < CTRLN_) ? Whead[(size_t)(n - OUT_) * H_ + k] : 0.f;
    Wo2[idx] = (__bf16)v;
}

__global__ void k_cvt_simple(const float* __restrict__ src, __bf16* __restrict__ dst, int n)
{
    int idx = blockIdx.x * blockDim.x + threadIdx.x;
    if (idx < n) dst[idx] = (__bf16)src[idx];
}

__global__ void k_prep_bias(const float* __restrict__ bih, const float* __restrict__ bhh,
                            const float* __restrict__ bout, const float* __restrict__ bhead,
                            float* __restrict__ bsum, float* __restrict__ bias2)
{
    int idx = blockIdx.x * blockDim.x + threadIdx.x;
    if (idx < G4H_) bsum[idx] = bih[idx] + bhh[idx];
    if (idx < N2_)
        bias2[idx] = (idx < OUT_) ? bout[idx]
                   : (idx < CTRLN_) ? bhead[idx - OUT_] : 0.f;
}

__global__ void k_init(const float* __restrict__ h0, const float* __restrict__ c0,
                       float* __restrict__ c, __bf16* __restrict__ xh)
{
    int idx = blockIdx.x * blockDim.x + threadIdx.x;
    if (idx >= B_ * H_) return;
    int b = idx / H_, h = idx % H_;
    c[idx] = c0[idx];
    xh[(size_t)b * KIN_ + D_ + h] = (__bf16)h0[idx];
}

__global__ void k_xcopy(const __bf16* __restrict__ xbf, __bf16* __restrict__ xh, int t)
{
    int idx = blockIdx.x * blockDim.x + threadIdx.x;
    if (idx >= B_ * D_) return;
    int b = idx / D_, d = idx % D_;
    xh[(size_t)b * KIN_ + d] = xbf[((size_t)b * T_ + t) * D_ + d];
}

// ---------------------------------------------------------------------------
static inline int gemm_grid(int Mdim, int Ndim)
{
    int tiles = ((Mdim + 15) / 16) * ((Ndim + 15) / 16);
    return (tiles + 7) / 8;
}

extern "C" void kernel_launch(void* const* d_in, const int* in_sizes, int n_in,
                              void* d_out, int out_size, void* d_ws, size_t ws_size,
                              hipStream_t stream)
{
    const float* x     = (const float*)d_in[0];
    const float* h0    = (const float*)d_in[1];
    const float* c0    = (const float*)d_in[2];
    const float* mem0  = (const float*)d_in[3];
    const float* Wih   = (const float*)d_in[4];
    const float* Whh   = (const float*)d_in[5];
    const float* bih   = (const float*)d_in[6];
    const float* bhh   = (const float*)d_in[7];
    const float* Wout  = (const float*)d_in[8];
    const float* bout  = (const float*)d_in[9];
    const float* Wrd   = (const float*)d_in[10];
    const float* brd   = (const float*)d_in[11];
    const float* Whead = (const float*)d_in[12];
    const float* bhead = (const float*)d_in[13];
    // d_in[14] = K (== 8, compile-time KTOP_)

    char* base = (char*)d_ws;
    size_t off = 0;
    auto alloc = [&](size_t bytes) -> char* {
        char* p = base + off;
        off = (off + bytes + 255) & ~(size_t)255;
        return p;
    };
    float*  mem    = (float*) alloc((size_t)B_ * S_ * M_ * 4);       // 64 MB
    __bf16* xbf    = (__bf16*)alloc((size_t)B_ * T_ * D_ * 2);
    __bf16* Wcat   = (__bf16*)alloc((size_t)G4H_ * KIN_ * 2);
    __bf16* Wo2    = (__bf16*)alloc((size_t)N2_ * H_ * 2);
    __bf16* Wrdb   = (__bf16*)alloc((size_t)OUT_ * OUT_ * 2);
    float*  bsum   = (float*) alloc((size_t)G4H_ * 4);
    float*  bias2  = (float*) alloc((size_t)N2_ * 4);
    __bf16* xh     = (__bf16*)alloc((size_t)B_ * KIN_ * 2);
    float*  cbuf   = (float*) alloc((size_t)B_ * H_ * 4);
    float*  gates  = (float*) alloc((size_t)B_ * G4H_ * 4);
    float*  out2   = (float*) alloc((size_t)B_ * N2_ * 4);
    __bf16* keys   = (__bf16*)alloc((size_t)B_ * 16 * M_ * 2);
    float*  erase  = (float*) alloc((size_t)B_ * M_ * 4);
    float*  addv   = (float*) alloc((size_t)B_ * M_ * 4);
    float*  scores = (float*) alloc((size_t)B_ * 5 * S_ * 4);        // 5.2 MB
    __bf16* readsb = (__bf16*)alloc((size_t)B_ * R_ * M_ * 2);
    (void)ws_size; (void)n_in; (void)in_sizes; (void)out_size;

    float* outp = (float*)d_out;                                     // [B, T, OUT]

    // ----- one-time prep (per launch; deterministic) -----
    (void)hipMemcpyAsync(mem, mem0, (size_t)B_ * S_ * M_ * 4,
                         hipMemcpyDeviceToDevice, stream);
    k_cvt_wcat  <<<(G4H_ * KIN_ + 255) / 256, 256, 0, stream>>>(Wih, Whh, Wcat);
    k_cvt_wo2   <<<(N2_ * H_ + 255) / 256, 256, 0, stream>>>(Wout, Whead, Wo2);
    k_cvt_simple<<<(OUT_ * OUT_ + 255) / 256, 256, 0, stream>>>(Wrd, Wrdb, OUT_ * OUT_);
    k_cvt_simple<<<(B_ * T_ * D_ + 255) / 256, 256, 0, stream>>>(x, xbf, B_ * T_ * D_);
    k_prep_bias <<<(G4H_ + 255) / 256, 256, 0, stream>>>(bih, bhh, bout, bhead, bsum, bias2);
    k_init      <<<(B_ * H_ + 255) / 256, 256, 0, stream>>>(h0, c0, cbuf, xh);

    // ----- sequential scan over T steps -----
    for (int t = 0; t < T_; ++t) {
        k_xcopy<<<(B_ * D_ + 255) / 256, 256, 0, stream>>>(xbf, xh, t);

        // gates = [x|h] @ Wcat^T + (b_ih+b_hh)   (LDS-staged panel, 16x64/wave)
        k_gates_gemm<<<32, 256, 0, stream>>>(xh, Wcat, bsum, gates);

        k_cell<<<(B_ * H_ + 255) / 256, 256, 0, stream>>>(gates, cbuf, xh);

        // out2 = h @ [W_out|W_head]^T + [b_out|b_head]   (vt | ctrl)
        wmma_gemm_nt<<<gemm_grid(B_, N2_), 256, 0, stream>>>(
            xh + D_, KIN_, Wo2, H_, bias2, nullptr, 0, out2, N2_, B_, N2_, H_);

        k_heads <<<B_, 320, 0, stream>>>(out2, keys, erase, addv);
        k_scores<<<(B_ * (S_ / 16) + 7) / 8, 256, 0, stream>>>(mem, keys, scores);
        k_write <<<B_, 256, 0, stream>>>(mem, scores, keys, erase, addv);
        k_read  <<<B_ * R_, 256, 0, stream>>>(mem, scores, readsb);

        // out[:, t, :] = vt + reads @ W_rd^T + b_rd
        wmma_gemm_nt<<<gemm_grid(B_, OUT_), 256, 0, stream>>>(
            readsb, R_ * M_, Wrdb, R_ * M_, brd, out2, N2_,
            outp + (size_t)t * OUT_, T_ * OUT_, B_, OUT_, R_ * M_);
    }
}
